// GPSConv_19765439496825
// MI455X (gfx1250) — matrix operand order repaired and checked by reading, not compiled, not run on hardware
//
#include <hip/hip_runtime.h>
#include <hip/hip_bf16.h>

// ---------------------------------------------------------------------------
// Types matching CDNA5 WMMA builtin signatures
// ---------------------------------------------------------------------------
typedef _Float16 h8_t  __attribute__((ext_vector_type(8)));
typedef _Float16 v16h  __attribute__((ext_vector_type(16)));
typedef float    v8f   __attribute__((ext_vector_type(8)));

#define BB 64
#define LL 512
#define CC 256
#define HHEADS 8
#define NNODES (BB*LL)                  // 32768
#define NCX ((size_t)NNODES*CC)         // 8388608 elements

static __device__ __forceinline__ v16h cat16(h8_t lo, h8_t hi) {
  return __builtin_shufflevector(lo, hi, 0,1,2,3,4,5,6,7,8,9,10,11,12,13,14,15);
}

// ---------------------------------------------------------------------------
// Generic tiled WMMA GEMM:  out[M,N] = act( A[M,K](f16) @ W[N,K]^T(f16) + bias
//                                           (+ resid[M,N] f32) )
// Block: 256 threads = 8 waves as 2(M) x 4(N); each wave owns a 32x32 tile
// (2 A-frags x 2 B-frags -> 4 WMMAs per K-step, 2x operand reuse).
// grid = (M/64, N/128).  Requires M%64==0, N%128==0, K%32==0.
// A-frag layout (16-bit A 16x32, wave32): lane<16 -> K 0..15,
// lane>=16 -> K 16..31 (contiguous, as B).  B-frag: lane holds column N0+(lane&15),
// K = 16*(lane>>4)+0..15 contiguous.
// ---------------------------------------------------------------------------
template<bool RELU, bool OUTF16, bool RESID>
__global__ __launch_bounds__(256) void gemm_wmma(
    const _Float16* __restrict__ A, const _Float16* __restrict__ W,
    const float* __restrict__ bias, const float* __restrict__ resid,
    void* __restrict__ outp, int M, int N, int K)
{
  const int w    = threadIdx.x >> 5;
  const int lane = threadIdx.x & 31;
  const int hf   = lane >> 4;
  const int l15  = lane & 15;
  const int wm   = w >> 2;
  const int wn   = w & 3;
  const int M0   = blockIdx.x * 64 + wm * 32;
  const int N0   = blockIdx.y * 128 + wn * 32;

  v8f a00, a01, a10, a11;
#pragma unroll
  for (int r = 0; r < 8; ++r) { a00[r] = 0.f; a01[r] = 0.f; a10[r] = 0.f; a11[r] = 0.f; }

  const _Float16* arow0 = A + (size_t)(M0 + l15) * K;
  const _Float16* arow1 = A + (size_t)(M0 + 16 + l15) * K;
  const _Float16* wrow0 = W + (size_t)(N0 + l15) * K;
  const _Float16* wrow1 = W + (size_t)(N0 + 16 + l15) * K;

  for (int k0 = 0; k0 < K; k0 += 32) {
    v16h af0 = *(const v16h*)(arow0 + k0 + 16 * hf);
    v16h af1 = *(const v16h*)(arow1 + k0 + 16 * hf);
    v16h bf0 = *(const v16h*)(wrow0 + k0 + 16 * hf);
    v16h bf1 = *(const v16h*)(wrow1 + k0 + 16 * hf);
    a00 = __builtin_amdgcn_wmma_f32_16x16x32_f16(false, af0, false, bf0, (short)0, a00, false, false);
    a01 = __builtin_amdgcn_wmma_f32_16x16x32_f16(false, af0, false, bf1, (short)0, a01, false, false);
    a10 = __builtin_amdgcn_wmma_f32_16x16x32_f16(false, af1, false, bf0, (short)0, a10, false, false);
    a11 = __builtin_amdgcn_wmma_f32_16x16x32_f16(false, af1, false, bf1, (short)0, a11, false, false);
  }

  const float bv0 = bias[N0 + l15];
  const float bv1 = bias[N0 + 16 + l15];
  float*    of = (float*)outp;
  _Float16* oh = (_Float16*)outp;

#pragma unroll
  for (int mt = 0; mt < 2; ++mt) {
#pragma unroll
    for (int nt = 0; nt < 2; ++nt) {
      const v8f& acc = (mt == 0) ? (nt == 0 ? a00 : a01) : (nt == 0 ? a10 : a11);
      const int n = N0 + nt * 16 + l15;
      const float bv = (nt == 0) ? bv0 : bv1;
#pragma unroll
      for (int r = 0; r < 8; ++r) {
        const int row = M0 + mt * 16 + r + 8 * hf;
        float v = acc[r] + bv;
        if (RESID) v += resid[(size_t)row * N + n];
        if (RELU)  v = fmaxf(v, 0.f);
        const size_t o = (size_t)row * N + n;
        if (OUTF16) oh[o] = (_Float16)v; else of[o] = v;
      }
    }
  }
}

// ---------------------------------------------------------------------------
// f32 -> f16 convert
// ---------------------------------------------------------------------------
__global__ void cvt_f32_f16(const float* __restrict__ in,
                            _Float16* __restrict__ out, int n)
{
  int i = blockIdx.x * 256 + threadIdx.x;
  if (i < n) out[i] = (_Float16)in[i];
}

// ---------------------------------------------------------------------------
// k_per[b] = clamp(ceil(tr*L),1,512); meta[0] = max(k_per)
// ---------------------------------------------------------------------------
__global__ void kper_kernel(const float* __restrict__ tr,
                            int* __restrict__ kper, int* __restrict__ meta)
{
  __shared__ int red[BB];
  int i = threadIdx.x;                      // blockDim = 64
  int k = (int)ceilf(tr[i] * (float)LL);
  k = min(max(k, 1), LL);
  kper[i] = k;
  red[i] = k;
  __syncthreads();
  for (int s = 32; s > 0; s >>= 1) {
    if (i < s) red[i] = max(red[i], red[i + s]);
    __syncthreads();
  }
  if (i == 0) meta[0] = red[0];
}

// ---------------------------------------------------------------------------
// scores[row] = dot(s1h[row,:], Ws2) + bs2 ; one wave per row
// grid = 4096 blocks * 8 waves
// ---------------------------------------------------------------------------
__global__ __launch_bounds__(256) void scores_kernel(
    const _Float16* __restrict__ s1h, const float* __restrict__ Ws2,
    const float* __restrict__ bs2, float* __restrict__ scores)
{
  int w = threadIdx.x >> 5, lane = threadIdx.x & 31;
  int row = blockIdx.x * 8 + w;
  const _Float16* a = s1h + (size_t)row * CC;
  float acc = 0.f;
  for (int i = lane; i < CC; i += 32) acc += (float)a[i] * Ws2[i];
  for (int off = 16; off > 0; off >>= 1) acc += __shfl_xor(acc, off, 32);
  if (lane == 0) scores[row] = acc + bs2[0];
}

// ---------------------------------------------------------------------------
// y_soft = softmax(log_softmax(scores) + gumbel) per batch row of 512
// block = 256 threads, each owns cols t and t+256
// ---------------------------------------------------------------------------
__global__ void softmax_kernel(const float* __restrict__ scores,
                               const float* __restrict__ gum,
                               float* __restrict__ ysoft)
{
  __shared__ float red[256];
  int b = blockIdx.x, t = threadIdx.x;
  const float* s = scores + b * LL;
  const float* g = gum + b * LL;
  float v0 = s[t], v1 = s[t + 256];

  float mx = fmaxf(v0, v1);
  red[t] = mx; __syncthreads();
  for (int st = 128; st; st >>= 1) { if (t < st) red[t] = fmaxf(red[t], red[t + st]); __syncthreads(); }
  mx = red[0]; __syncthreads();

  red[t] = expf(v0 - mx) + expf(v1 - mx); __syncthreads();
  for (int st = 128; st; st >>= 1) { if (t < st) red[t] += red[t + st]; __syncthreads(); }
  float lse = mx + logf(red[0]); __syncthreads();

  float z0 = v0 - lse + g[t];
  float z1 = v1 - lse + g[t + 256];

  float mz = fmaxf(z0, z1);
  red[t] = mz; __syncthreads();
  for (int st = 128; st; st >>= 1) { if (t < st) red[t] = fmaxf(red[t], red[t + st]); __syncthreads(); }
  mz = red[0]; __syncthreads();

  float e0 = expf(z0 - mz), e1 = expf(z1 - mz);
  red[t] = e0 + e1; __syncthreads();
  for (int st = 128; st; st >>= 1) { if (t < st) red[t] += red[t + st]; __syncthreads(); }
  float inv = 1.f / red[0];
  ysoft[b * LL + t]       = e0 * inv;
  ysoft[b * LL + t + 256] = e1 * inv;
}

// ---------------------------------------------------------------------------
// Top-k selection reproducing jax.lax.top_k tie semantics (lower index wins).
// slot s<k_per -> s-th smallest kept index; slots [k_per,k_max) -> smallest
// indices with refined==0.
// ---------------------------------------------------------------------------
__global__ void select_kernel(const float* __restrict__ ysoft,
                              const int* __restrict__ kperArr,
                              const int* __restrict__ meta,
                              int* __restrict__ topk)
{
  __shared__ float y[LL];
  __shared__ int   kept[LL];
  __shared__ int   refn[LL];
  int b = blockIdx.x, t = threadIdx.x;
  y[t]       = ysoft[b * LL + t];
  y[t + 256] = ysoft[b * LL + t + 256];
  __syncthreads();
  int kmax = meta[0], kp = kperArr[b];
  int jj[2] = {t, t + 256};
  int pk[2];

  for (int u = 0; u < 2; ++u) {
    int j = jj[u]; float yj = y[j]; int rank = 0;
    for (int i = 0; i < LL; ++i) {
      float yi = y[i];
      rank += (yi > yj) || (yi == yj && i < j);
    }
    kept[j] = (rank < kmax) ? 1 : 0;
  }
  __syncthreads();
  for (int u = 0; u < 2; ++u) {
    int j = jj[u]; int c = 0;
    for (int i = 0; i < j; ++i) c += kept[i];
    pk[u] = c;
    refn[j] = (kept[j] && c < kp) ? 1 : 0;
  }
  __syncthreads();
  for (int u = 0; u < 2; ++u) {
    int j = jj[u];
    if (refn[j]) {
      topk[b * LL + pk[u]] = j;
    } else {
      int z = 0;
      for (int i = 0; i < j; ++i) z += (refn[i] == 0);
      if (z < kmax - kp) topk[b * LL + kp + z] = j;
    }
  }
}

// ---------------------------------------------------------------------------
// Gather compact tokens: h_compact[b,s,:] = x_f16[node] if s<k_per else 0
// grid = B*512 blocks, 256 threads (channel)
// ---------------------------------------------------------------------------
__global__ void gather_kernel(const _Float16* __restrict__ xh,
                              const int* __restrict__ topk,
                              const int* __restrict__ kperArr,
                              _Float16* __restrict__ hcmp)
{
  int bs = blockIdx.x;
  int b = bs >> 9, s = bs & 511, c = threadIdx.x;
  _Float16 v = (_Float16)0.f;
  if (s < kperArr[b]) {
    int j = topk[b * LL + s];
    v = xh[((size_t)(b * LL + j)) * CC + c];
  }
  hcmp[(size_t)bs * CC + c] = v;
}

// ---------------------------------------------------------------------------
// Fused attention per (b, head, 64-query tile), two-pass exact softmax.
// DH = 32 so each 16x16 S tile is a single 16x16x32 WMMA.
// Pass 1: Q.K^T WMMAs with *unmasked raw* running max per lane (softmax is
//         shift-invariant and compact K-rows >= k_per are exactly zero, so
//         raw max >= valid max and exp(s-m) <= 1 is guaranteed) -> inner loop
//         is just 2 WMMA + 8 v_max3.  One butterfly reduction at the end,
//         then scaled by 1/sqrt(DH).
// Pass 2: recompute S (WMMA is the abundant resource), p = ok? exp(s*sc-m):0,
//         per-lane partial row sums, P routed via LDS (C-layout -> A-layout),
//         O accumulated with WMMA against V^T staged in LDS.
// Block = 4 waves (16 q rows each).
// ---------------------------------------------------------------------------
__global__ __launch_bounds__(128) void attn_kernel(
    const _Float16* __restrict__ qkv,   // [B*512, 768]  q|k|v
    _Float16* __restrict__ hattn,       // [B*512, 256]
    const int* __restrict__ kperArr, const int* __restrict__ meta)
{
  const int b   = blockIdx.x;
  const int hid = blockIdx.y;
  const int qt  = blockIdx.z;
  const int kmax = meta[0];
  if (qt * 64 >= kmax) return;
  const int kp = kperArr[b];

  const int tid = threadIdx.x, w = tid >> 5, lane = tid & 31;
  const int hf = lane >> 4, l15 = lane & 15;

  __shared__ __align__(32) _Float16 Vt[32][512];     // V^T for this (b,head)
  __shared__ __align__(32) _Float16 Pb[4][16][32];   // per-wave P staging

  // stage V^T: Vt[dh][key] = v[b,key,hid*32+dh]
  for (int idx = tid; idx < 512 * 32; idx += 128) {
    int key = idx >> 5, dh = idx & 31;
    Vt[dh][key] = qkv[((size_t)b * LL + key) * 768 + 512 + hid * 32 + dh];
  }
  __syncthreads();

  const int qbase = qt * 64 + w * 16;
  const size_t qrow = (size_t)b * LL + qbase + l15;
  const _Float16* qp = qkv + qrow * 768 + hid * 32;
  v16h qfrag = *(const v16h*)(qp + 16 * hf);

  const float sc = 0.17677669529663687f;   // 1/sqrt(32)
  const _Float16* kbase = qkv + (size_t)b * LL * 768 + 256 + hid * 32 + 16 * hf;

  // -------- pass 1: raw running row max (no masking, no scaling in-loop) ---
  float m[8];
#pragma unroll
  for (int r = 0; r < 8; ++r) m[r] = -3e38f;

  for (int ks = 0; ks < 512; ks += 32) {
    v16h kf0 = *(const v16h*)(kbase + (size_t)(ks + l15) * 768);
    v16h kf1 = *(const v16h*)(kbase + (size_t)(ks + 16 + l15) * 768);
    v8f sa, sb;
#pragma unroll
    for (int r = 0; r < 8; ++r) { sa[r] = 0.f; sb[r] = 0.f; }
    sa = __builtin_amdgcn_wmma_f32_16x16x32_f16(false, qfrag, false, kf0, (short)0, sa, false, false);
    sb = __builtin_amdgcn_wmma_f32_16x16x32_f16(false, qfrag, false, kf1, (short)0, sb, false, false);
#pragma unroll
    for (int r = 0; r < 8; ++r) m[r] = fmaxf(m[r], fmaxf(sa[r], sb[r]));
  }
  // one butterfly reduction per row (within each 16-lane group), then scale
#pragma unroll
  for (int r = 0; r < 8; ++r) {
    for (int off = 1; off < 16; off <<= 1) m[r] = fmaxf(m[r], __shfl_xor(m[r], off, 32));
    m[r] *= sc;        // shift m' >= true valid max (rows >= k_per are zero)
  }

  // -------- pass 2: p = exp(s*sc - m), per-lane partial sums, O via WMMA ---
  v8f o0, o1;
  float lsum[8];
#pragma unroll
  for (int r = 0; r < 8; ++r) { o0[r] = 0.f; o1[r] = 0.f; lsum[r] = 0.f; }

  for (int ks = 0; ks < 512; ks += 32) {
    v16h kf0 = *(const v16h*)(kbase + (size_t)(ks + l15) * 768);
    v16h kf1 = *(const v16h*)(kbase + (size_t)(ks + 16 + l15) * 768);
    v8f sa, sb;
#pragma unroll
    for (int r = 0; r < 8; ++r) { sa[r] = 0.f; sb[r] = 0.f; }
    sa = __builtin_amdgcn_wmma_f32_16x16x32_f16(false, qfrag, false, kf0, (short)0, sa, false, false);
    sb = __builtin_amdgcn_wmma_f32_16x16x32_f16(false, qfrag, false, kf1, (short)0, sb, false, false);
    const bool va_ok = (ks + l15) < kp, vb_ok = (ks + 16 + l15) < kp;
#pragma unroll
    for (int r = 0; r < 8; ++r) {
      float pa = va_ok ? expf(fmaf(sa[r], sc, -m[r])) : 0.f;
      float pb = vb_ok ? expf(fmaf(sb[r], sc, -m[r])) : 0.f;
      lsum[r] += pa + pb;               // per-lane partial; reduced once at end
      Pb[w][r + 8 * hf][l15]      = (_Float16)pa;
      Pb[w][r + 8 * hf][l15 + 16] = (_Float16)pb;
    }
    __syncthreads();

    const _Float16* pp = &Pb[w][l15][0];
    v16h pf = *(const v16h*)(pp + 16 * hf);
    v16h bv0 = *(const v16h*)(&Vt[l15][ks + 16 * hf]);
    v16h bv1 = *(const v16h*)(&Vt[16 + l15][ks + 16 * hf]);
    o0 = __builtin_amdgcn_wmma_f32_16x16x32_f16(false, pf, false, bv0, (short)0, o0, false, false);
    o1 = __builtin_amdgcn_wmma_f32_16x16x32_f16(false, pf, false, bv1, (short)0, o1, false, false);
    __syncthreads();
  }

  // final sum reduction + normalize + store
#pragma unroll
  for (int r = 0; r < 8; ++r) {
    float s = lsum[r];
    for (int off = 1; off < 16; off <<= 1) s += __shfl_xor(s, off, 32);
    float inv = 1.f / s;
    int q = qbase + r + 8 * hf;
    size_t base = ((size_t)b * LL + q) * CC + hid * 32;
    hattn[base + l15]      = (_Float16)(o0[r] * inv);
    hattn[base + 16 + l15] = (_Float16)(o1[r] * inv);
  }
}

// ---------------------------------------------------------------------------
// h = x * ratio for all nodes (overwritten by scatter for selected ones)
// ---------------------------------------------------------------------------
__global__ void init_h_kernel(const float* __restrict__ x,
                              const float* __restrict__ tr,
                              float* __restrict__ h, _Float16* __restrict__ hh)
{
  size_t idx = (size_t)blockIdx.x * 256 + threadIdx.x;
  int n = (int)(idx >> 8);
  int b = n >> 9;
  float v = x[idx] * tr[b];
  h[idx] = v;
  hh[idx] = (_Float16)v;
}

// ---------------------------------------------------------------------------
// scatter: for slot s<k_max, node=topk[b,s]: h[node]=(x[node]+proj[b,s])*r
// grid = (B, 512), 256 threads (channel)
// ---------------------------------------------------------------------------
__global__ void scatter_kernel(const float* __restrict__ x,
                               const float* __restrict__ tr,
                               const float* __restrict__ hproj,
                               const int* __restrict__ topk,
                               const int* __restrict__ meta,
                               float* __restrict__ h, _Float16* __restrict__ hh)
{
  int b = blockIdx.x, s = blockIdx.y;
  if (s >= meta[0]) return;
  int j = topk[b * LL + s];
  int c = threadIdx.x;
  size_t node = (size_t)b * LL + j;
  float v = (x[node * CC + c] + hproj[((size_t)b * LL + s) * CC + c]) * tr[b];
  h[node * CC + c] = v;
  hh[node * CC + c] = (_Float16)v;
}

// ---------------------------------------------------------------------------
// Host orchestration
// ---------------------------------------------------------------------------
extern "C" void kernel_launch(void* const* d_in, const int* in_sizes, int n_in,
                              void* d_out, int out_size, void* d_ws, size_t ws_size,
                              hipStream_t stream)
{
  const float* x    = (const float*)d_in[0];
  const float* tr   = (const float*)d_in[3];
  const float* gum  = (const float*)d_in[4];
  const float* Ws1  = (const float*)d_in[5];
  const float* bs1  = (const float*)d_in[6];
  const float* Ws2  = (const float*)d_in[7];
  const float* bs2  = (const float*)d_in[8];
  const float* Win  = (const float*)d_in[9];
  const float* b_in = (const float*)d_in[10];
  const float* Wout = (const float*)d_in[11];
  const float* bout = (const float*)d_in[12];
  const float* Wm1  = (const float*)d_in[13];
  const float* bm1  = (const float*)d_in[14];
  const float* Wm2  = (const float*)d_in[15];
  const float* bm2  = (const float*)d_in[16];

  char* p = (char*)d_ws;
  auto alloc = [&](size_t bytes) -> char* {
    char* r = p; p += (bytes + 255) & ~(size_t)255; return r;
  };

  _Float16* xh    = (_Float16*)alloc(NCX * 2);
  _Float16* Ws1h  = (_Float16*)alloc((size_t)CC * CC * 2);
  _Float16* Winh  = (_Float16*)alloc((size_t)3 * CC * CC * 2);
  _Float16* Wouth = (_Float16*)alloc((size_t)CC * CC * 2);
  _Float16* Wm1h  = (_Float16*)alloc((size_t)2 * CC * CC * 2);
  _Float16* Wm2h  = (_Float16*)alloc((size_t)2 * CC * CC * 2);
  _Float16* s1h   = (_Float16*)alloc(NCX * 2);
  float*    scr   = (float*)alloc((size_t)BB * LL * 4);
  float*    ysoft = (float*)alloc((size_t)BB * LL * 4);
  int*      kper  = (int*)alloc((size_t)BB * 4);
  int*      meta  = (int*)alloc(256);
  int*      topk  = (int*)alloc((size_t)BB * LL * 4);
  _Float16* hcmp  = (_Float16*)alloc(NCX * 2);
  _Float16* qkvh  = (_Float16*)alloc((size_t)NNODES * 768 * 2);
  _Float16* hattn = (_Float16*)alloc(NCX * 2);
  float*    hproj = (float*)alloc(NCX * 4);
  float*    hbuf  = (float*)alloc(NCX * 4);
  _Float16* hhalf = (_Float16*)alloc(NCX * 2);
  _Float16* m1h   = (_Float16*)alloc((size_t)NNODES * 512 * 2);

  // 1) converts
  cvt_f32_f16<<<(int)((NCX + 255) / 256), 256, 0, stream>>>(x, xh, (int)NCX);
  cvt_f32_f16<<<(65536 + 255) / 256, 256, 0, stream>>>(Ws1, Ws1h, 65536);
  cvt_f32_f16<<<(196608 + 255) / 256, 256, 0, stream>>>(Win, Winh, 196608);
  cvt_f32_f16<<<(65536 + 255) / 256, 256, 0, stream>>>(Wout, Wouth, 65536);
  cvt_f32_f16<<<(131072 + 255) / 256, 256, 0, stream>>>(Wm1, Wm1h, 131072);
  cvt_f32_f16<<<(131072 + 255) / 256, 256, 0, stream>>>(Wm2, Wm2h, 131072);

  // 2) k_per / k_max
  kper_kernel<<<1, 64, 0, stream>>>(tr, kper, meta);

  // 3) scorer hidden: s1 = relu(x @ Ws1^T + bs1)
  gemm_wmma<true, true, false><<<dim3(NNODES / 64, CC / 128), 256, 0, stream>>>(
      xh, Ws1h, bs1, nullptr, s1h, NNODES, CC, CC);

  // 4) scores + 5) y_soft
  scores_kernel<<<NNODES / 8, 256, 0, stream>>>(s1h, Ws2, bs2, scr);
  softmax_kernel<<<BB, 256, 0, stream>>>(scr, gum, ysoft);

  // 6) top-k selection
  select_kernel<<<BB, 256, 0, stream>>>(ysoft, kper, meta, topk);

  // 7) gather compact tokens
  gather_kernel<<<BB * LL, 256, 0, stream>>>(xh, topk, kper, hcmp);

  // 8) qkv = h_compact @ Win^T + b_in
  gemm_wmma<false, true, false><<<dim3(NNODES / 64, (3 * CC) / 128), 256, 0, stream>>>(
      hcmp, Winh, b_in, nullptr, qkvh, NNODES, 3 * CC, CC);

  // 9) fused attention (two-pass exact softmax)
  attn_kernel<<<dim3(BB, HHEADS, LL / 64), 128, 0, stream>>>(qkvh, hattn, kper, meta);

  // 10) out-proj (f32 out)
  gemm_wmma<false, false, false><<<dim3(NNODES / 64, CC / 128), 256, 0, stream>>>(
      hattn, Wouth, bout, nullptr, hproj, NNODES, CC, CC);

  // 11) residual init + 12) scatter
  init_h_kernel<<<(int)(NCX / 256), 256, 0, stream>>>(x, tr, hbuf, hhalf);
  scatter_kernel<<<dim3(BB, LL), 256, 0, stream>>>(x, tr, hproj, topk, meta, hbuf, hhalf);

  // 13) MLP layer 1: m1 = relu(h @ Wm1^T + bm1)
  gemm_wmma<true, true, false><<<dim3(NNODES / 64, (2 * CC) / 128), 256, 0, stream>>>(
      hhalf, Wm1h, bm1, nullptr, m1h, NNODES, 2 * CC, CC);

  // 14) out = h + m1 @ Wm2^T + bm2  (f32 into d_out)
  gemm_wmma<false, false, true><<<dim3(NNODES / 64, CC / 128), 256, 0, stream>>>(
      m1h, Wm2h, bm2, hbuf, d_out, NNODES, CC, 2 * CC);

  (void)in_sizes; (void)n_in; (void)out_size; (void)ws_size;
}
